// GATEncoder_81638738363133
// MI455X (gfx1250) — matrix operand order repaired and checked by reading, not compiled, not run on hardware
//
#include <hip/hip_runtime.h>
#include <hip/hip_bf16.h>
#include <math.h>

typedef __attribute__((ext_vector_type(16))) _Float16 v16h;
typedef __attribute__((ext_vector_type(8)))  float    v8f;

#define IN_DIM  32
#define HID     64
#define HEADS   4
#define HC      256
#define OUT_DIM 128

__device__ __forceinline__ float elu_f(float x) { return x > 0.f ? x : (__expf(x) - 1.f); }

// order-preserving float<->int map for atomicMax-based segment max
__device__ __forceinline__ int   f2ord(float f) { int i = __float_as_int(f); return i < 0 ? (i ^ 0x7fffffff) : i; }
__device__ __forceinline__ float ord2f(int i)   { return __int_as_float(i < 0 ? (i ^ 0x7fffffff) : i); }

__device__ __forceinline__ void edge_sd(const int* __restrict__ ei, int E, int ed, int& s, int& d) {
    if (ed < E) { s = ei[ed]; d = ei[E + ed]; } else { s = d = ed - E; }
}

// ---------------------------------------------------------------------------
// Weight pre-pack: B[K,Nc] f32 row-major -> fragment-major f16.
// Per (k-chunk of 32, n-tile of 16): 512 f16, lane-major (lane*16 + e), where
// lane holds col = n0+(lane&15), k = k0 + (lane>>4)*16 + e.  Each GEMM wave
// then fetches its whole B fragment as one aligned 32-byte vector load.
// ---------------------------------------------------------------------------
__global__ void pack_b_f16(const float* __restrict__ B, _Float16* __restrict__ Bp,
                           int K, int Nc)
{
    int id = blockIdx.x * blockDim.x + threadIdx.x;
    int total = K * Nc;
    if (id >= total) return;
    int e    = id & 15;
    int lane = (id >> 4) & 31;
    int blk  = id >> 9;                 // (k-chunk, n-tile) block index
    int tilesN = Nc >> 4;
    int kc = blk / tilesN, nt = blk % tilesN;
    int k   = kc * 32 + (lane >> 4) * 16 + e;
    int col = nt * 16 + (lane & 15);
    Bp[id] = (_Float16)B[(size_t)k * Nc + col];
}

// ---------------------------------------------------------------------------
// GEMM: C[M,Nc] = act(A[M,K] @ B + bias), A f32 row-major (converted on load),
// B pre-packed f16 fragments. One wave per 16x16 C tile, dual WMMA chains.
// ---------------------------------------------------------------------------
template<bool HAS_BIAS, bool ACT_ELU>
__global__ __launch_bounds__(256)
void wmma_gemm(const float* __restrict__ A, const _Float16* __restrict__ Bp,
               const float* __restrict__ bias, float* __restrict__ C,
               int M, int K, int Nc)
{
    const int lane   = threadIdx.x & 31;
    const int wave   = threadIdx.x >> 5;
    const int tilesN = Nc >> 4;
    const int nTiles = (M >> 4) * tilesN;
    const int tile   = blockIdx.x * 8 + wave;
    if (tile >= nTiles) return;                 // wave-uniform: EXEC stays all-ones

    const int m0   = (tile / tilesN) << 4;
    const int nt   = tile % tilesN;
    const int n0   = nt << 4;
    const int half = lane >> 4;
    const int r15  = lane & 15;

    const float*    Arow = A + (size_t)(m0 + r15) * K;
    const _Float16* Bcol = Bp + ((size_t)nt << 9) + lane * 16;   // + kc*tilesN*512
    const size_t    Bstride = (size_t)tilesN << 9;               // per k-chunk

    // load one 16x32 A fragment (f32 -> f16) + one pre-packed B fragment
    auto load_a = [&](int k0) -> v16h {
        const float4* Ar = (const float4*)(Arow + k0);
        float4 p0 = Ar[half * 4 + 0];
        float4 p1 = Ar[half * 4 + 1];
        float4 p2 = Ar[half * 4 + 2];
        float4 p3 = Ar[half * 4 + 3];
        v16h a;
        a[0]=(_Float16)p0.x; a[1]=(_Float16)p0.y; a[2]=(_Float16)p0.z; a[3]=(_Float16)p0.w;
        a[4]=(_Float16)p1.x; a[5]=(_Float16)p1.y; a[6]=(_Float16)p1.z; a[7]=(_Float16)p1.w;
        a[8]=(_Float16)p2.x; a[9]=(_Float16)p2.y; a[10]=(_Float16)p2.z; a[11]=(_Float16)p2.w;
        a[12]=(_Float16)p3.x; a[13]=(_Float16)p3.y; a[14]=(_Float16)p3.z; a[15]=(_Float16)p3.w;
        return a;
    };
    auto load_b = [&](int kc) -> v16h {
        return *(const v16h*)(Bcol + (size_t)kc * Bstride);
    };

    v8f acc0 = {}, acc1 = {};
    int k0 = 0;
    for (; k0 + 64 <= K; k0 += 64) {
        __builtin_prefetch(Arow + k0 + 64, 0, 1);      // global_prefetch_b8
        v16h a0 = load_a(k0);
        v16h b0 = load_b(k0 >> 5);
        v16h a1 = load_a(k0 + 32);
        v16h b1 = load_b((k0 >> 5) + 1);
        acc0 = __builtin_amdgcn_wmma_f32_16x16x32_f16(false, a0, false, b0, (short)0, acc0, false, false);
        acc1 = __builtin_amdgcn_wmma_f32_16x16x32_f16(false, a1, false, b1, (short)0, acc1, false, false);
    }
    if (k0 < K) {                                      // K == 32 tail (in-proj)
        v16h a0 = load_a(k0);
        v16h b0 = load_b(k0 >> 5);
        acc0 = __builtin_amdgcn_wmma_f32_16x16x32_f16(false, a0, false, b0, (short)0, acc0, false, false);
    }

    float bv = HAS_BIAS ? bias[n0 + r15] : 0.f;
    #pragma unroll
    for (int r = 0; r < 8; ++r) {
        int row = m0 + half * 8 + r;
        float v = acc0[r] + acc1[r] + bv;
        if (ACT_ELU) v = elu_f(v);
        C[(size_t)row * Nc + n0 + r15] = v;
    }
}

// per-(node,head) attention scores
__global__ void att_scores(const float* __restrict__ hl,
                           const float* __restrict__ att_src,
                           const float* __restrict__ att_dst,
                           float* __restrict__ asrc, float* __restrict__ adst, int N)
{
    int id = blockIdx.x * blockDim.x + threadIdx.x;
    if (id >= N * HEADS) return;
    int n = id >> 2, h = id & 3;
    const float4* hp = (const float4*)(hl + (size_t)n * HC + h * HID);
    const float4* as = (const float4*)(att_src + h * HID);
    const float4* ad = (const float4*)(att_dst + h * HID);
    float s = 0.f, d = 0.f;
    #pragma unroll 4
    for (int c = 0; c < HID / 4; ++c) {
        float4 v = hp[c], a = as[c], b = ad[c];
        s += v.x * a.x + v.y * a.y + v.z * a.z + v.w * a.w;
        d += v.x * b.x + v.y * b.y + v.z * b.z + v.w * b.w;
    }
    asrc[id] = s; adst[id] = d;
}

// pass 1: e = leaky_relu(asrc[src]+adst[dst]); segment max (ordered-int atomicMax)
__global__ void edge_logits(const int* __restrict__ ei, int E, int Ep,
                            const float* __restrict__ asrc, const float* __restrict__ adst,
                            float* __restrict__ ebuf, int* __restrict__ mbuf)
{
    int id = blockIdx.x * blockDim.x + threadIdx.x;
    if (id >= Ep * HEADS) return;
    int ed = id >> 2, h = id & 3;
    int s, d; edge_sd(ei, E, ed, s, d);
    float e = asrc[s * HEADS + h] + adst[d * HEADS + h];
    e = e > 0.f ? e : 0.2f * e;
    ebuf[id] = e;
    atomicMax(mbuf + d * HEADS + h, f2ord(e));
}

// pass 2: p = exp(e - m[dst]); segment sum into denom
__global__ void edge_softmax_num(const int* __restrict__ ei, int E, int Ep,
                                 const int* __restrict__ mbuf,
                                 float* __restrict__ ebuf, float* __restrict__ denom)
{
    int id = blockIdx.x * blockDim.x + threadIdx.x;
    if (id >= Ep * HEADS) return;
    int ed = id >> 2, h = id & 3;
    int s, d; edge_sd(ei, E, ed, s, d);
    float p = __expf(ebuf[id] - ord2f(mbuf[d * HEADS + h]));
    ebuf[id] = p;
    atomicAdd(denom + d * HEADS + h, p);
}

// pass 3: one block per edge; thread t handles channel t of 256
__global__ __launch_bounds__(256)
void edge_scatter(const int* __restrict__ ei, int E, int Ep,
                  const float* __restrict__ hl, const float* __restrict__ ebuf,
                  const float* __restrict__ denom, float* __restrict__ acc)
{
    int ed = blockIdx.x;
    int t  = threadIdx.x;
    int h  = t >> 6;
    int s, d; edge_sd(ei, E, ed, s, d);
    float alpha = ebuf[ed * HEADS + h] / (denom[d * HEADS + h] + 1e-16f);
    atomicAdd(acc + (size_t)d * HC + t, hl[(size_t)s * HC + t] * alpha);
}

// h_next = elu(acc + bias)
__global__ void node_epilogue(const float* __restrict__ acc, const float* __restrict__ bias,
                              float* __restrict__ out, int N)
{
    int id = blockIdx.x * blockDim.x + threadIdx.x;
    if (id >= N * HC) return;
    out[id] = elu_f(acc[id] + bias[id & (HC - 1)]);
}

// layernorm over 128 cols; one wave32 per node
__global__ __launch_bounds__(256)
void layernorm_rows(const float* __restrict__ H, const float* __restrict__ gamma,
                    const float* __restrict__ beta, float* __restrict__ out, int N)
{
    int lane = threadIdx.x & 31;
    int wave = threadIdx.x >> 5;
    int n = blockIdx.x * 8 + wave;
    if (n >= N) return;
    const float* row = H + (size_t)n * OUT_DIM;
    float v[4], s = 0.f, ss = 0.f;
    #pragma unroll
    for (int i = 0; i < 4; ++i) { v[i] = row[lane + 32 * i]; s += v[i]; ss += v[i] * v[i]; }
    #pragma unroll
    for (int off = 16; off; off >>= 1) {
        s  += __shfl_xor(s,  off, 32);
        ss += __shfl_xor(ss, off, 32);
    }
    float mu  = s  * (1.f / OUT_DIM);
    float var = ss * (1.f / OUT_DIM) - mu * mu;
    float inv = rsqrtf(var + 1e-5f);
    float* orow = out + (size_t)n * OUT_DIM;
    #pragma unroll
    for (int i = 0; i < 4; ++i) {
        int c = lane + 32 * i;
        orow[c] = (v[i] - mu) * inv * gamma[c] + beta[c];
    }
}

extern "C" void kernel_launch(void* const* d_in, const int* in_sizes, int n_in,
                              void* d_out, int out_size, void* d_ws, size_t ws_size,
                              hipStream_t stream)
{
    (void)n_in; (void)out_size; (void)ws_size;
    const float* x     = (const float*)d_in[0];
    const int*   ei    = (const int*)  d_in[1];
    const float* W_in  = (const float*)d_in[2];
    const float* b_in  = (const float*)d_in[3];
    const float* lin0  = (const float*)d_in[4];
    const float* att0s = (const float*)d_in[5];
    const float* att0d = (const float*)d_in[6];
    const float* bias0 = (const float*)d_in[7];
    const float* lin1  = (const float*)d_in[8];
    const float* att1s = (const float*)d_in[9];
    const float* att1d = (const float*)d_in[10];
    const float* bias1 = (const float*)d_in[11];
    const float* W_out = (const float*)d_in[12];
    const float* b_out = (const float*)d_in[13];
    const float* gmm   = (const float*)d_in[14];
    const float* bta   = (const float*)d_in[15];
    float* out = (float*)d_out;

    const int N  = in_sizes[0] / IN_DIM;
    const int E  = in_sizes[1] / 2;
    const int Ep = E + N;

    // workspace carve-out
    float* ws = (float*)d_ws;
    size_t off = 0;
    float* h0    = ws + off; off += (size_t)N * HID;
    float* hl    = ws + off; off += (size_t)N * HC;
    float* acc   = ws + off; off += (size_t)N * HC;
    float* hin   = ws + off; off += (size_t)N * HC;
    float* asrc  = ws + off; off += (size_t)N * HEADS;
    float* adst  = ws + off; off += (size_t)N * HEADS;
    int*   mbuf  = (int*)(ws + off); off += (size_t)N * HEADS;
    float* denom = ws + off; off += (size_t)N * HEADS;
    float* ebuf  = ws + off; off += (size_t)Ep * HEADS;
    _Float16* wp = (_Float16*)(ws + off);
    _Float16* Wp_in  = wp; wp += IN_DIM * HID;
    _Float16* Wp_l0  = wp; wp += HID * HC;
    _Float16* Wp_l1  = wp; wp += HC * HC;
    _Float16* Wp_out = wp; wp += HC * OUT_DIM;

    // pre-pack all weights into WMMA fragment layout (f16)
    auto pack = [&](const float* B, _Float16* Bp, int K, int Nc) {
        int total = K * Nc;
        pack_b_f16<<<(total + 255) / 256, 256, 0, stream>>>(B, Bp, K, Nc);
    };
    pack(W_in,  Wp_in,  IN_DIM, HID);
    pack(lin0,  Wp_l0,  HID,    HC);
    pack(lin1,  Wp_l1,  HC,     HC);
    pack(W_out, Wp_out, HC,     OUT_DIM);

    auto nblk = [](int M, int Nc) { return ((M / 16) * (Nc / 16) + 7) / 8; };

    // input projection: h0 = elu(x @ W_in + b_in)
    wmma_gemm<true, true><<<nblk(N, HID), 256, 0, stream>>>(x, Wp_in, b_in, h0, N, IN_DIM, HID);

    for (int layer = 0; layer < 2; ++layer) {
        const float*    Ain = (layer == 0) ? h0    : hin;
        const int       Kin = (layer == 0) ? HID   : HC;
        const _Float16* lw  = (layer == 0) ? Wp_l0 : Wp_l1;
        const float*    ats = (layer == 0) ? att0s : att1s;
        const float*    atd = (layer == 0) ? att0d : att1d;
        const float*    bs  = (layer == 0) ? bias0 : bias1;

        (void)hipMemsetAsync(acc,   0,    (size_t)N * HC    * sizeof(float), stream);
        (void)hipMemsetAsync(denom, 0,    (size_t)N * HEADS * sizeof(float), stream);
        (void)hipMemsetAsync(mbuf,  0x80, (size_t)N * HEADS * sizeof(int),   stream);

        wmma_gemm<false, false><<<nblk(N, HC), 256, 0, stream>>>(Ain, lw, nullptr, hl, N, Kin, HC);

        int nh = N * HEADS;
        att_scores<<<(nh + 255) / 256, 256, 0, stream>>>(hl, ats, atd, asrc, adst, N);

        int eh = Ep * HEADS;
        edge_logits<<<(eh + 255) / 256, 256, 0, stream>>>(ei, E, Ep, asrc, adst, ebuf, mbuf);
        edge_softmax_num<<<(eh + 255) / 256, 256, 0, stream>>>(ei, E, Ep, mbuf, ebuf, denom);
        edge_scatter<<<Ep, 256, 0, stream>>>(ei, E, Ep, hl, ebuf, denom, acc);

        node_epilogue<<<(N * HC + 255) / 256, 256, 0, stream>>>(acc, bs, hin, N);
    }

    // output projection + layernorm
    wmma_gemm<true, false><<<nblk(N, OUT_DIM), 256, 0, stream>>>(hin, Wp_out, b_out, hl, N, HC, OUT_DIM);
    layernorm_rows<<<(N + 7) / 8, 256, 0, stream>>>(hl, gmm, bta, out, N);
}
